// MoEVectorized_82592221102586
// MI455X (gfx1250) — matrix operand order repaired and checked by reading, not compiled, not run on hardware
//
#include <hip/hip_runtime.h>

// MoE top-2 FFN for MI455X (gfx1250): bf16 WMMA GEMMs, fp32 accumulate,
// fused bias + LeakyReLU + top-2 gating, deterministic gating-loss reduction.
// M-tile = 32 tokens/workgroup so every weight (B) fragment feeds 2 WMMAs,
// halving L2 weight traffic vs a 16-row tile.

static constexpr int Dd = 512;    // model dim
static constexpr int Hh = 1024;   // ffn hidden
static constexpr int Ee = 8;      // experts
static constexpr int Nn = 16384;  // tokens (B*T)
#define NEG_SLOPE 0.01f
#define HPAD 1032                 // 1024 + 8 elem pad -> conflict-free LDS rows

typedef __attribute__((ext_vector_type(8)))  __bf16 bf16x8;
typedef __attribute__((ext_vector_type(16))) __bf16 v16bf;
typedef __attribute__((ext_vector_type(8)))  float  v8f;

__device__ __forceinline__ unsigned short f32_bf16(float f) {
  unsigned u = __float_as_uint(f);
  u += 0x7FFFu + ((u >> 16) & 1u);            // round-to-nearest-even
  return (unsigned short)(u >> 16);
}

__device__ __forceinline__ v16bf mk16(bf16x8 lo, bf16x8 hi) {
  return __builtin_shufflevector(lo, hi, 0,1,2,3,4,5,6,7,8,9,10,11,12,13,14,15);
}

__device__ __forceinline__ float leaky(float v) {
  return (v >= 0.0f) ? v : NEG_SLOPE * v;
}

// ---------------- precision-conversion kernels ----------------

__global__ void cvt_f32_bf16(const float* __restrict__ in,
                             unsigned short* __restrict__ out, int n) {
  int i = blockIdx.x * blockDim.x + threadIdx.x;
  if (i < n) out[i] = f32_bf16(in[i]);
}

// in: [E][R][C] fp32  ->  out: [E][C][R] bf16   (coalesced writes)
__global__ void cvt_transpose(const float* __restrict__ in,
                              unsigned short* __restrict__ out, int R, int C) {
  int i = blockIdx.x * blockDim.x + threadIdx.x;
  int total = Ee * R * C;
  if (i >= total) return;
  int e = i / (R * C);
  int r2 = i - e * (R * C);
  int c = r2 / R;
  int r = r2 - c * R;
  out[i] = f32_bf16(in[(e * R + r) * C + c]);
}

// ---------------- router: softmax + top-2 gates + prob partial sums ----------------

__global__ void router_kernel(const float* __restrict__ x,
                              const float* __restrict__ Wr,
                              const float* __restrict__ br,
                              float* __restrict__ gates,
                              float* __restrict__ partial) {
  __shared__ float red[256];
  int n = blockIdx.x * blockDim.x + threadIdx.x;   // 64 blocks * 256 = 16384
  float lg[Ee];
  #pragma unroll
  for (int e = 0; e < Ee; e++) lg[e] = br[e];
  const float* xr = x + n * Dd;
  for (int d = 0; d < Dd; d++) {
    float xv = xr[d];
    #pragma unroll
    for (int e = 0; e < Ee; e++) lg[e] = fmaf(xv, Wr[d * Ee + e], lg[e]);
  }
  float m = lg[0];
  #pragma unroll
  for (int e = 1; e < Ee; e++) m = fmaxf(m, lg[e]);
  float p[Ee]; float s = 0.0f;
  #pragma unroll
  for (int e = 0; e < Ee; e++) { p[e] = __expf(lg[e] - m); s += p[e]; }
  float inv = 1.0f / s;
  #pragma unroll
  for (int e = 0; e < Ee; e++) p[e] *= inv;

  // top-2 (ties -> lowest index, matching lax.top_k)
  int i1 = 0;
  #pragma unroll
  for (int e = 1; e < Ee; e++) if (p[e] > p[i1]) i1 = e;
  int i2 = (i1 == 0) ? 1 : 0;
  #pragma unroll
  for (int e = 0; e < Ee; e++) if (e != i1 && p[e] > p[i2]) i2 = e;

  #pragma unroll
  for (int e = 0; e < Ee; e++)
    gates[n * Ee + e] = (e == i1) ? p[i1] : ((e == i2) ? p[i2] : 0.0f);

  // deterministic per-block reduction of router probs
  for (int e = 0; e < Ee; e++) {
    red[threadIdx.x] = p[e];
    __syncthreads();
    for (int st = 128; st > 0; st >>= 1) {
      if (threadIdx.x < st) red[threadIdx.x] += red[threadIdx.x + st];
      __syncthreads();
    }
    if (threadIdx.x == 0) partial[blockIdx.x * Ee + e] = red[0];
    __syncthreads();
  }
}

__global__ void finalize_kernel(const float* __restrict__ partial,
                                float* __restrict__ loss_out) {
  if (threadIdx.x == 0 && blockIdx.x == 0) {
    float loss = 0.0f;
    for (int e = 0; e < Ee; e++) {
      float s = 0.0f;
      for (int b = 0; b < Nn / 256; b++) s += partial[b * Ee + e];
      float d = (1.0f / Ee) - s / (float)Nn;
      loss += d * d;
    }
    loss_out[0] = (loss / (float)Ee) * 1e-4f;
  }
}

// ---------------- fused two-layer expert FFN with WMMA ----------------
// grid: Nn/32 blocks of 256 threads (8 waves). Each block: one 32-token tile
// (two 16-row WMMA row-tiles sharing every B fragment).
// wave w: phase1 -> h cols [w*128, w*128+128); phase2 -> out cols [w*64, w*64+64).

__global__ __launch_bounds__(256, 1)
void moe_ffn_kernel(const unsigned short* __restrict__ xbv,
                    const unsigned short* __restrict__ w1tv,
                    const unsigned short* __restrict__ w2tv,
                    const float* __restrict__ b1,
                    const float* __restrict__ b2,
                    const float* __restrict__ gates,
                    float* __restrict__ out) {
  __shared__ unsigned short hT[32 * HPAD];   // bf16 h tile (32 rows), padded
  __shared__ float sg[32 * Ee];
  __shared__ unsigned smask;

  const int tid   = threadIdx.x;
  const int wave  = tid >> 5;
  const int lane  = tid & 31;
  const int lhalf = lane >> 4;      // which 16-lane half
  const int l16   = lane & 15;
  const int kLoA  = lhalf * 8;      // A frag: K split {0-7,16-23} vs {8-15,24-31}
  const int kLoB  = lhalf * 8;      // B frag: K split {0-7,16-23} vs {8-15,24-31}
  const int tile0 = blockIdx.x * 32;

  const __bf16* xb  = (const __bf16*)xbv;
  const __bf16* w1t = (const __bf16*)w1tv;
  const __bf16* w2t = (const __bf16*)w2tv;

  if (tid < 32 * Ee) {
    int t = tid >> 3, e = tid & 7;
    sg[t * Ee + e] = gates[(tile0 + t) * Ee + e];
  }
  __syncthreads();
  if (tid == 0) {
    unsigned m = 0;
    for (int e = 0; e < Ee; e++) {
      float a = 0.0f;
      for (int t = 0; t < 32; t++) a += sg[t * Ee + e];
      if (a != 0.0f) m |= 1u << e;
    }
    smask = m;
  }
  __syncthreads();
  const unsigned mask = smask;

  v8f outacc[2][4];
  #pragma unroll
  for (int rt = 0; rt < 2; rt++)
    #pragma unroll
    for (int c = 0; c < 4; c++)
      outacc[rt][c] = v8f{0.f,0.f,0.f,0.f,0.f,0.f,0.f,0.f};

  const __bf16* xrow0 = xb + (tile0 + l16) * Dd;
  const __bf16* xrow1 = xb + (tile0 + 16 + l16) * Dd;

  for (int e = 0; e < Ee; e++) {
    if (!((mask >> e) & 1u)) continue;   // uniform across the block

    // ---- phase 1: h = leaky(x @ W1[e] + b1[e]) ----
    v8f acc[2][8];
    #pragma unroll
    for (int rt = 0; rt < 2; rt++)
      #pragma unroll
      for (int c = 0; c < 8; c++)
        acc[rt][c] = v8f{0.f,0.f,0.f,0.f,0.f,0.f,0.f,0.f};

    #pragma unroll 1
    for (int k = 0; k < Dd; k += 32) {
      v16bf A0 = mk16(*(const bf16x8*)(xrow0 + k + kLoA),
                      *(const bf16x8*)(xrow0 + k + 16 + kLoA));
      v16bf A1 = mk16(*(const bf16x8*)(xrow1 + k + kLoA),
                      *(const bf16x8*)(xrow1 + k + 16 + kLoA));
      #pragma unroll
      for (int c = 0; c < 8; c++) {
        int ncol = wave * 128 + c * 16 + l16;
        const __bf16* brow = w1t + (e * Hh + ncol) * Dd;
        v16bf B = mk16(*(const bf16x8*)(brow + k + kLoB),
                       *(const bf16x8*)(brow + k + 16 + kLoB));
        acc[0][c] = __builtin_amdgcn_wmma_f32_16x16x32_bf16(
            false, A0, false, B, (short)0, acc[0][c], false, false);
        acc[1][c] = __builtin_amdgcn_wmma_f32_16x16x32_bf16(
            false, A1, false, B, (short)0, acc[1][c], false, false);
      }
    }
    // epilogue: bias + leaky, bf16 into LDS h tile
    #pragma unroll
    for (int c = 0; c < 8; c++) {
      int ncol = wave * 128 + c * 16 + l16;
      float bias = b1[e * Hh + ncol];
      #pragma unroll
      for (int rt = 0; rt < 2; rt++) {
        #pragma unroll
        for (int j = 0; j < 8; j++) {
          int row = rt * 16 + j + lhalf * 8;     // C layout: VGPR j -> rows j / j+8
          hT[row * HPAD + ncol] = f32_bf16(leaky(acc[rt][c][j] + bias));
        }
      }
    }
    __syncthreads();

    // ---- phase 2: y = leaky(h @ W2[e] + b2[e]); outacc += g * y ----
    v8f p2[2][4];
    #pragma unroll
    for (int rt = 0; rt < 2; rt++)
      #pragma unroll
      for (int c = 0; c < 4; c++)
        p2[rt][c] = v8f{0.f,0.f,0.f,0.f,0.f,0.f,0.f,0.f};

    const __bf16* hrow0 = ((const __bf16*)hT) + l16 * HPAD;
    const __bf16* hrow1 = ((const __bf16*)hT) + (16 + l16) * HPAD;
    #pragma unroll 1
    for (int k = 0; k < Hh; k += 32) {
      v16bf A0 = mk16(*(const bf16x8*)(hrow0 + k + kLoA),
                      *(const bf16x8*)(hrow0 + k + 16 + kLoA));
      v16bf A1 = mk16(*(const bf16x8*)(hrow1 + k + kLoA),
                      *(const bf16x8*)(hrow1 + k + 16 + kLoA));
      #pragma unroll
      for (int c = 0; c < 4; c++) {
        int ncol = wave * 64 + c * 16 + l16;
        const __bf16* brow = w2t + (e * Dd + ncol) * Hh;
        v16bf B = mk16(*(const bf16x8*)(brow + k + kLoB),
                       *(const bf16x8*)(brow + k + 16 + kLoB));
        p2[0][c] = __builtin_amdgcn_wmma_f32_16x16x32_bf16(
            false, A0, false, B, (short)0, p2[0][c], false, false);
        p2[1][c] = __builtin_amdgcn_wmma_f32_16x16x32_bf16(
            false, A1, false, B, (short)0, p2[1][c], false, false);
      }
    }
    #pragma unroll
    for (int rt = 0; rt < 2; rt++) {
      #pragma unroll
      for (int j = 0; j < 8; j++) {
        int row = rt * 16 + j + lhalf * 8;
        float g = sg[row * Ee + e];
        #pragma unroll
        for (int c = 0; c < 4; c++) {
          int ncol = wave * 64 + c * 16 + l16;
          float v = leaky(p2[rt][c][j] + b2[e * Dd + ncol]);
          outacc[rt][c][j] = fmaf(g, v, outacc[rt][c][j]);
        }
      }
    }
    __syncthreads();   // before next expert overwrites hT
  }

  // ---- store output tile ----
  #pragma unroll
  for (int rt = 0; rt < 2; rt++) {
    #pragma unroll
    for (int c = 0; c < 4; c++) {
      int ncol = wave * 64 + c * 16 + l16;
      #pragma unroll
      for (int j = 0; j < 8; j++) {
        int row = rt * 16 + j + lhalf * 8;
        out[(tile0 + row) * Dd + ncol] = outacc[rt][c][j];
      }
    }
  }
}

// ---------------- launch ----------------

extern "C" void kernel_launch(void* const* d_in, const int* in_sizes, int n_in,
                              void* d_out, int out_size, void* d_ws, size_t ws_size,
                              hipStream_t stream) {
  const float* x  = (const float*)d_in[0];   // [N, D]
  const float* Wr = (const float*)d_in[1];   // [D, E]
  const float* br = (const float*)d_in[2];   // [E]
  const float* W1 = (const float*)d_in[3];   // [E, D, H]
  const float* b1 = (const float*)d_in[4];   // [E, H]
  const float* W2 = (const float*)d_in[5];   // [E, H, D]
  const float* b2 = (const float*)d_in[6];   // [E, D]
  float* out = (float*)d_out;                // [N*D] output, then 1 loss scalar

  char* ws = (char*)d_ws;
  unsigned short* xb   = (unsigned short*)(ws);                 // N*D bf16   (16 MiB)
  unsigned short* w1t  = (unsigned short*)(ws + 16777216);      // E*H*D bf16 ( 8 MiB)
  unsigned short* w2t  = (unsigned short*)(ws + 25165824);      // E*D*H bf16 ( 8 MiB)
  float* gates         = (float*)(ws + 33554432);               // N*E f32    (512 KiB)
  float* partial       = (float*)(ws + 34078720);               // 64*E f32

  cvt_f32_bf16 <<<(Nn * Dd) / 256, 256, 0, stream>>>(x, xb, Nn * Dd);
  cvt_transpose<<<(Ee * Dd * Hh) / 256, 256, 0, stream>>>(W1, w1t, Dd, Hh); // -> [E][H][D]
  cvt_transpose<<<(Ee * Hh * Dd) / 256, 256, 0, stream>>>(W2, w2t, Hh, Dd); // -> [E][D][H]
  router_kernel<<<Nn / 256, 256, 0, stream>>>(x, Wr, br, gates, partial);
  moe_ffn_kernel<<<Nn / 32, 256, 0, stream>>>(xb, w1t, w2t, b1, b2, gates, out);
  finalize_kernel<<<1, 32, 0, stream>>>(partial, out + (size_t)Nn * Dd);
}
